// DragonCurveAttention_66425964200588
// MI455X (gfx1250) — matrix operand order repaired and checked by reading, not compiled, not run on hardware
//
#include <hip/hip_runtime.h>
#include <hip/hip_bf16.h>

typedef __attribute__((ext_vector_type(4)))  unsigned int u32x4;
typedef __attribute__((ext_vector_type(8)))  int          i32x8;
typedef __attribute__((ext_vector_type(4)))  int          i32x4;
typedef __attribute__((ext_vector_type(16))) __bf16       v16bf;
typedef __attribute__((ext_vector_type(8)))  float        v8f;
typedef __attribute__((ext_vector_type(4)))  float        f32x4;
typedef __attribute__((ext_vector_type(4)))  __bf16       bf16x4;

#define B_DIM 8
#define S_DIM 2048
#define D_DIM 512

union FragU {
  v16bf bf;
  u32x4 q[2];
};

static __device__ inline v8f wmma_bf16(v16bf a, v16bf b, v8f c) {
  return __builtin_amdgcn_wmma_f32_16x16x32_bf16(false, a, false, b, (short)0, c,
                                                 false, false);
}

// ---------------------------------------------------------------------------
// Kernel 1: build dragon weights (softplus(affine(pattern))) * 1/sqrt(D),
//           plus the scalar blend factor from the tiny MLP.  One block.
// ---------------------------------------------------------------------------
__global__ void dragon_prep_kernel(const float* __restrict__ pscale,
                                   const float* __restrict__ pbias,
                                   const float* __restrict__ w1,
                                   const float* __restrict__ b1,
                                   const float* __restrict__ w2,
                                   const float* __restrict__ b2,
                                   float* __restrict__ pa,       // 4096 scratch
                                   float* __restrict__ pb,       // 4096 scratch
                                   float* __restrict__ dragon_w, // 2048
                                   float* __restrict__ scal)     // [blend, cu]
{
  __shared__ float rmin[256];
  __shared__ float rmax[256];
  const int tid = threadIdx.x;

  if (tid == 0) pa[0] = 0.5f;
  __syncthreads();

  int len = 1;
  for (int it = 1; it < 12; ++it) {
    const int nl = 2 * len + 1;
    // expansion: [prev, 0.5, 1 - reverse(prev)]
    for (int i = tid; i < nl; i += 256) {
      float v;
      if (i < len)       v = pa[i];
      else if (i == len) v = 0.5f;
      else               v = 1.0f - pa[len - 1 - (i - len - 1)];
      pb[i] = v;
    }
    __syncthreads();
    // conv1d kernel [1/3,1/3,1/3], zero-padded 'same'
    for (int i = tid; i < nl; i += 256) {
      float a = pb[i];
      if (i > 0)      a += pb[i - 1];
      if (i + 1 < nl) a += pb[i + 1];
      pa[i] = a * (1.0f / 3.0f);
    }
    __syncthreads();
    // min/max normalize
    float mn = 1e30f, mx = -1e30f;
    for (int i = tid; i < nl; i += 256) {
      mn = fminf(mn, pa[i]);
      mx = fmaxf(mx, pa[i]);
    }
    rmin[tid] = mn; rmax[tid] = mx;
    __syncthreads();
    for (int s = 128; s > 0; s >>= 1) {
      if (tid < s) {
        rmin[tid] = fminf(rmin[tid], rmin[tid + s]);
        rmax[tid] = fmaxf(rmax[tid], rmax[tid + s]);
      }
      __syncthreads();
    }
    mn = rmin[0]; mx = rmax[0];
    const float inv = 1.0f / (mx - mn + 1e-8f);
    for (int i = tid; i < nl; i += 256) pa[i] = (pa[i] - mn) * inv;
    __syncthreads();
    len = nl;
  }

  // it = min(log2(2048), 11) = 11 ; len = 4095 ; interp hits pattern[2s] exactly
  const float sc = pscale[11];
  const float bs = pbias[11];
  const float dscale = rsqrtf((float)D_DIM);  // fold 1/sqrt(D) into key weights
  for (int s = tid; s < S_DIM; s += 256) {
    const float x  = pa[2 * s] * sc + bs;
    const float sp = (x > 20.0f) ? x : log1pf(__expf(x));
    dragon_w[s] = sp * dscale;
  }

  if (tid == 0) {
    float acc = b2[0];
    for (int j = 0; j < 64; ++j) {
      float h = w1[11 * 64 + j] + b1[j];   // one_hot(11) @ w1 + b1
      h = fmaxf(h, 0.0f);
      acc += h * w2[j];
    }
    const float bl = 1.0f / (1.0f + __expf(-acc));
    scal[0] = bl;
    scal[1] = (1.0f - bl) / (float)S_DIM;
  }
}

// ---------------------------------------------------------------------------
// Kernel 2: kw = bf16(K * dragon_w[s])   (scale already folded into dragon_w)
// ---------------------------------------------------------------------------
__global__ void build_kw_kernel(const float* __restrict__ k,
                                const float* __restrict__ dw,
                                __bf16* __restrict__ kw)
{
  const size_t i  = (size_t)blockIdx.x * 256 + threadIdx.x;
  const size_t i4 = i * 4;
  const int    s  = (int)((i4 >> 9) & (S_DIM - 1));  // D=512 -> >>9
  const f32x4  v  = *(const f32x4*)(k + i4);
  const float  w  = dw[s];
  bf16x4 o;
  o.x = (__bf16)(v.x * w);
  o.y = (__bf16)(v.y * w);
  o.z = (__bf16)(v.z * w);
  o.w = (__bf16)(v.w * w);
  *(bf16x4*)(kw + i4) = o;
}

// ---------------------------------------------------------------------------
// Kernel 2b: qw = bf16(Q)  (source tensor for the TDM Q-tile DMA)
// ---------------------------------------------------------------------------
__global__ void build_qw_kernel(const float* __restrict__ q,
                                __bf16* __restrict__ qw)
{
  const size_t i4 = ((size_t)blockIdx.x * 256 + threadIdx.x) * 4;
  const f32x4  v  = *(const f32x4*)(q + i4);
  bf16x4 o;
  o.x = (__bf16)v.x;
  o.y = (__bf16)v.y;
  o.z = (__bf16)v.z;
  o.w = (__bf16)v.w;
  *(bf16x4*)(qw + i4) = o;
}

// ---------------------------------------------------------------------------
// Kernel 3: vt[b][d][s] = bf16(V[b][s][d])  (32x32 tiled transpose)
// ---------------------------------------------------------------------------
__global__ void transpose_v_kernel(const float* __restrict__ v,
                                   __bf16* __restrict__ vt)
{
  __shared__ float tile[32][33];
  const int b  = blockIdx.z;
  const int d0 = blockIdx.x * 32;
  const int s0 = blockIdx.y * 32;
  const int tx = threadIdx.x;   // 32
  const int ty = threadIdx.y;   // 8

  for (int j = ty; j < 32; j += 8)
    tile[j][tx] = v[((size_t)b * S_DIM + s0 + j) * D_DIM + d0 + tx];
  __syncthreads();
  for (int j = ty; j < 32; j += 8)
    vt[((size_t)b * D_DIM + d0 + j) * S_DIM + s0 + tx] = (__bf16)tile[tx][j];
}

// ---------------------------------------------------------------------------
// Kernel 4a/4b: deterministic column-sum of V over s (two stages, no atomics)
// ---------------------------------------------------------------------------
__global__ void colsum_part_kernel(const float* __restrict__ v,
                                   float* __restrict__ vpart)
{
  const int d  = blockIdx.x * 256 + threadIdx.x;
  const int b  = blockIdx.y;
  const int sp = blockIdx.z;            // 16 splits of 128 keys
  float a = 0.0f;
  const int s0 = sp * 128;
  for (int s = 0; s < 128; ++s)
    a += v[((size_t)b * S_DIM + s0 + s) * D_DIM + d];
  vpart[((size_t)b * 16 + sp) * D_DIM + d] = a;
}

__global__ void colsum_final_kernel(const float* __restrict__ vpart,
                                    float* __restrict__ vsum)
{
  const int d = blockIdx.x * 256 + threadIdx.x;
  const int b = blockIdx.y;
  float a = 0.0f;
  for (int sp = 0; sp < 16; ++sp)
    a += vpart[((size_t)b * 16 + sp) * D_DIM + d];
  vsum[(size_t)b * D_DIM + d] = a;
}

// ---------------------------------------------------------------------------
// Kernel 5: fused flash attention with bf16 WMMA.
//   Workgroup = 8 waves, 16 query rows. Flash tile = 256 keys.
//   Q tile is DMA'd into LDS by the Tensor Data Mover (TENSORcnt tracked).
//   Wave w: QK^T for its 32-key slice (full D contraction), shared softmax
//   stats in LDS, P staged in LDS (A-layout friendly), then P@V over its
//   64-output-dim slice.
// ---------------------------------------------------------------------------
__global__ __launch_bounds__(256)
void flash_attn_kernel(const __bf16* __restrict__ qw,   // [B][S][D] bf16
                       const __bf16* __restrict__ kw,   // [B][S][D] bf16
                       const __bf16* __restrict__ vt,   // [B][D][S] bf16
                       const float* __restrict__ vsum,  // [B][D]
                       const float* __restrict__ scal,  // blend, (1-blend)/S
                       float* __restrict__ out)
{
  __shared__ alignas(16) __bf16 Qlds[16 * D_DIM];   // 16 KB
  __shared__ alignas(16) __bf16 Plds[16 * 256];     // 8 KB
  __shared__ float redmax[8 * 16];
  __shared__ float redsum[8 * 16];

  const int tid  = threadIdx.x;
  const int wv   = tid >> 5;
  const int lane = tid & 31;
  const int l16  = lane & 15;
  const int hi   = lane >> 4;        // 0: lanes 0-15, 1: lanes 16-31
  const int b    = blockIdx.y;
  const int q0   = blockIdx.x * 16;

  // ---- TDM: DMA the 16x512 bf16 Q tile from global into LDS (wave 0) ------
  if (tid < 32) {
    const __bf16* gq = qw + ((size_t)b * S_DIM + q0) * D_DIM;
    const unsigned long long ga = (unsigned long long)(uintptr_t)gq;
    const unsigned int lds_addr = (unsigned int)(uintptr_t)(&Qlds[0]);

    u32x4 g0;
    g0[0] = 1u;                                   // count=1, user descriptor
    g0[1] = lds_addr;                             // lds_addr @ bits 63:32
    g0[2] = (unsigned int)(ga & 0xFFFFFFFFu);     // global_addr[31:0]
    g0[3] = (unsigned int)((ga >> 32) & 0x01FFFFFFu)  // global_addr[56:32]
            | (2u << 30);                         // type = 2 ("image")

    i32x8 g1;
    g1[0] = (int)(1u << 16);                      // wg_mask=0, data_size=1 (2B)
    g1[1] = (int)(((unsigned)D_DIM & 0xFFFFu) << 16); // tensor_dim0 @ bit48
    g1[2] = (int)(((unsigned)D_DIM >> 16) | (16u << 16)); // dim0 hi | tensor_dim1=16
    g1[3] = (int)((unsigned)D_DIM << 16);         // tile_dim0=512 @ bit112
    g1[4] = 16;                                   // tile_dim1=16, tile_dim2=0
    g1[5] = D_DIM;                                // tensor_dim0_stride=512
    g1[6] = 0;                                    // stride hi / dim1_stride
    g1[7] = 0;
    i32x4 gz4 = (i32x4){0, 0, 0, 0};
    i32x8 gz8 = (i32x8){0, 0, 0, 0, 0, 0, 0, 0};

    // 6-arg form (clang-23 / therock-10.0 headers)
    __builtin_amdgcn_tensor_load_to_lds(g0, g1, gz4, gz4, gz8, 0);
    __builtin_amdgcn_s_wait_tensorcnt(0);
  }
  __syncthreads();

  v8f acc[4];
#pragma unroll
  for (int c = 0; c < 4; ++c) acc[c] = (v8f){0,0,0,0,0,0,0,0};

  float m_run[8], l_run[8];
#pragma unroll
  for (int r = 0; r < 8; ++r) { m_run[r] = -1e30f; l_run[r] = 0.0f; }

  const float blend = scal[0];
  const float cu    = scal[1];

  const __bf16* kw_b = kw + (size_t)b * S_DIM * D_DIM;
  const __bf16* vt_b = vt + (size_t)b * D_DIM * S_DIM;

  for (int tb = 0; tb < S_DIM; tb += 256) {
    // -------- GEMM 1: S_w = Q x Kw^T for keys [tb + 32*wv, +32) ------------
    v8f s0 = (v8f){0,0,0,0,0,0,0,0};
    v8f s1 = (v8f){0,0,0,0,0,0,0,0};
    const int key0 = tb + wv * 32;
#pragma unroll
    for (int k0 = 0; k0 < D_DIM; k0 += 32) {
      FragU a;
      const __bf16* qp = &Qlds[l16 * D_DIM + k0 + hi * 8];
      a.q[0] = *(const u32x4*)qp;
      a.q[1] = *(const u32x4*)(qp + 16);

      FragU kb0, kb1;
      const __bf16* kp0 = kw_b + (size_t)(key0 + l16) * D_DIM + k0 + hi * 8;
      const __bf16* kp1 = kw_b + (size_t)(key0 + 16 + l16) * D_DIM + k0 + hi * 8;
      kb0.q[0] = *(const u32x4*)kp0;
      kb0.q[1] = *(const u32x4*)(kp0 + 16);
      kb1.q[0] = *(const u32x4*)kp1;
      kb1.q[1] = *(const u32x4*)(kp1 + 16);

      s0 = wmma_bf16(a.bf, kb0.bf, s0);
      s1 = wmma_bf16(a.bf, kb1.bf, s1);
    }

    // -------- partial row-max (within wave, across the 16-lane half) -------
    float pm[8];
#pragma unroll
    for (int r = 0; r < 8; ++r) pm[r] = fmaxf(s0[r], s1[r]);
#pragma unroll
    for (int off = 1; off < 16; off <<= 1) {
#pragma unroll
      for (int r = 0; r < 8; ++r)
        pm[r] = fmaxf(pm[r], __shfl_xor(pm[r], off, 32));
    }
    if (l16 == 0) {
#pragma unroll
      for (int r = 0; r < 8; ++r) redmax[wv * 16 + hi * 8 + r] = pm[r];
    }
    __syncthreads();

    // -------- global row-max, correction, P = exp(S - m_new) ---------------
    float corr[8], ps[8];
#pragma unroll
    for (int r = 0; r < 8; ++r) {
      float g = m_run[r];
#pragma unroll
      for (int w = 0; w < 8; ++w) g = fmaxf(g, redmax[w * 16 + hi * 8 + r]);
      corr[r]  = __expf(m_run[r] - g);
      m_run[r] = g;
      const float p0 = __expf(s0[r] - g);
      const float p1 = __expf(s1[r] - g);
      ps[r] = p0 + p1;
      const int row = r + hi * 8;
      Plds[row * 256 + wv * 32 + l16]      = (__bf16)p0;
      Plds[row * 256 + wv * 32 + 16 + l16] = (__bf16)p1;
    }
#pragma unroll
    for (int off = 1; off < 16; off <<= 1) {
#pragma unroll
      for (int r = 0; r < 8; ++r) ps[r] += __shfl_xor(ps[r], off, 32);
    }
    if (l16 == 0) {
#pragma unroll
      for (int r = 0; r < 8; ++r) redsum[wv * 16 + hi * 8 + r] = ps[r];
    }
    __syncthreads();

#pragma unroll
    for (int r = 0; r < 8; ++r) {
      float t = 0.0f;
#pragma unroll
      for (int w = 0; w < 8; ++w) t += redsum[w * 16 + hi * 8 + r];
      l_run[r] = l_run[r] * corr[r] + t;
    }

    // rescale O accumulators
#pragma unroll
    for (int c = 0; c < 4; ++c) {
#pragma unroll
      for (int r = 0; r < 8; ++r) acc[c][r] *= corr[r];
    }

    // prefetch next tile's K rows (speculative, L2)
    if (tb + 256 < S_DIM)
      __builtin_prefetch(kw_b + (size_t)(tb + 256 + wv * 32 + l16) * D_DIM, 0, 1);

    // -------- GEMM 2: O += P x V  (256 keys, wave's 64 output dims) --------
#pragma unroll
    for (int k0 = 0; k0 < 256; k0 += 32) {
      FragU pfrag;
      const __bf16* pp = &Plds[l16 * 256 + k0 + hi * 8];
      pfrag.q[0] = *(const u32x4*)pp;
      pfrag.q[1] = *(const u32x4*)(pp + 16);
#pragma unroll
      for (int c = 0; c < 4; ++c) {
        const int dim = wv * 64 + c * 16 + l16;
        const __bf16* vp = vt_b + (size_t)dim * S_DIM + tb + k0 + hi * 8;
        FragU vb;
        vb.q[0] = *(const u32x4*)vp;
        vb.q[1] = *(const u32x4*)(vp + 16);
        acc[c] = wmma_bf16(pfrag.bf, vb.bf, acc[c]);
      }
    }
    __syncthreads();   // protect Plds / redmax / redsum for next iteration
  }

  // -------- epilogue: out = blend * O/l + cu * colsum(V) -------------------
#pragma unroll
  for (int c = 0; c < 4; ++c) {
    const int dim = wv * 64 + c * 16 + l16;
    const float vs = vsum[(size_t)b * D_DIM + dim];
#pragma unroll
    for (int r = 0; r < 8; ++r) {
      const int row = r + hi * 8;
      const float val = blend * (acc[c][r] / l_run[r]) + cu * vs;
      out[((size_t)b * S_DIM + q0 + row) * D_DIM + dim] = val;
    }
  }
}

// ---------------------------------------------------------------------------
// Launch
// ---------------------------------------------------------------------------
extern "C" void kernel_launch(void* const* d_in, const int* in_sizes, int n_in,
                              void* d_out, int out_size, void* d_ws, size_t ws_size,
                              hipStream_t stream) {
  const float* q      = (const float*)d_in[0];
  const float* k      = (const float*)d_in[1];
  const float* v      = (const float*)d_in[2];
  const float* pscale = (const float*)d_in[3];
  const float* pbias  = (const float*)d_in[4];
  const float* w1     = (const float*)d_in[5];
  const float* b1     = (const float*)d_in[6];
  const float* w2     = (const float*)d_in[7];
  const float* b2     = (const float*)d_in[8];

  const size_t tensor_bytes = (size_t)B_DIM * S_DIM * D_DIM * 2;  // 16 MB bf16

  char* ws = (char*)d_ws;
  float* dragon_w = (float*)(ws + 0);            // 2048 f32
  float* pa       = (float*)(ws + 8192);         // 4096 f32
  float* pb       = (float*)(ws + 24576);        // 4096 f32
  float* vsum     = (float*)(ws + 40960);        // 8*512 f32
  float* scal     = (float*)(ws + 57344);        // 2 f32
  float* vpart    = (float*)(ws + 65536);        // 8*16*512 f32 = 256 KB
  __bf16* kw      = (__bf16*)(ws + 327680);                      // 16 MB
  __bf16* vt      = (__bf16*)(ws + 327680 + tensor_bytes);       // 16 MB
  __bf16* qw      = (__bf16*)(ws + 327680 + 2 * tensor_bytes);   // 16 MB

  dragon_prep_kernel<<<1, 256, 0, stream>>>(pscale, pbias, w1, b1, w2, b2,
                                            pa, pb, dragon_w, scal);

  const int nvec = (B_DIM * S_DIM * D_DIM / 4 + 255) / 256;  // 8192
  build_kw_kernel<<<nvec, 256, 0, stream>>>(k, dragon_w, kw);
  build_qw_kernel<<<nvec, 256, 0, stream>>>(q, qw);

  transpose_v_kernel<<<dim3(D_DIM / 32, S_DIM / 32, B_DIM), dim3(32, 8), 0,
                       stream>>>(v, vt);

  colsum_part_kernel<<<dim3(D_DIM / 256, B_DIM, 16), 256, 0, stream>>>(v, vpart);
  colsum_final_kernel<<<dim3(D_DIM / 256, B_DIM), 256, 0, stream>>>(vpart, vsum);

  flash_attn_kernel<<<dim3(S_DIM / 16, B_DIM), 256, 0, stream>>>(
      qw, kw, vt, vsum, scal, (float*)d_out);
}
